// EmbfeatureContextEncodingTransformer_18030272709048
// MI455X (gfx1250) — matrix operand order repaired and checked by reading, not compiled, not run on hardware
//
#include <hip/hip_runtime.h>
#include <cstddef>

// ---------------- types ----------------
typedef __attribute__((ext_vector_type(8)))  float  v8f;
typedef __attribute__((ext_vector_type(4)))  float  v4f;
typedef __attribute__((ext_vector_type(16))) __bf16 v16bf;
typedef __attribute__((ext_vector_type(8)))  __bf16 v8bf;

#define BT   32
#define DCH  768
#define PSP  1204   // 28*43
#define NFC  256
#define NFB  1024
#define NN   12
#define RR   384    // BT*NN
#define KSTEPS (DCH / 32)   // 24

// ---------------- helpers ----------------
__device__ __forceinline__ __bf16 to_bf16(float f) {
    union { float f; unsigned u; } in; in.f = f;
    unsigned u = in.u;
    unsigned r = u + 0x7FFFu + ((u >> 16) & 1u);   // round-to-nearest-even
    unsigned short h = (unsigned short)(r >> 16);
    union { unsigned short s; __bf16 b; } out; out.s = h;
    return out.b;
}

__device__ __forceinline__ v16bf cat16(v8bf lo, v8bf hi) {
    v16bf r;
#pragma unroll
    for (int e = 0; e < 8; ++e) { r[e] = lo[e]; r[e + 8] = hi[e]; }
    return r;
}

__device__ __forceinline__ v8f wmma_bf16(v16bf a, v16bf b, v8f c) {
    // (neg_a, A, neg_b, B, c_mod, C, reuse_a, reuse_b)
    return __builtin_amdgcn_wmma_f32_16x16x32_bf16(false, a, false, b, (short)0, c, false, false);
}

// A fragment 16x32: lane<16 -> K runs [k0,k0+16); lane>=16 -> [k0+16,k0+32)
__device__ __forceinline__ v16bf load_a_frag(const __bf16* base, int row, int rs, int k0, int lane) {
    int kb = k0 + ((lane & 16) ? 16 : 0);
    const __bf16* p = base + (size_t)row * rs + kb;
    v8bf lo = *(const v8bf*)p;
    v8bf hi = *(const v8bf*)(p + 8);
    return cat16(lo, hi);
}

// B fragment 32x16 for B = W^T, W row-major [ncols][rs] (W[n][k]):
// lane: N = n0 + lane%16, K = k0 + (lane<16?0:16) + e  (contiguous 16)
__device__ __forceinline__ v16bf load_bt_frag(const __bf16* W, int n0, int rs, int k0, int lane) {
    int kb = k0 + ((lane & 16) ? 16 : 0);
    const __bf16* p = W + (size_t)(n0 + (lane & 15)) * rs + kb;
    v8bf lo = *(const v8bf*)p;
    v8bf hi = *(const v8bf*)(p + 8);
    return cat16(lo, hi);
}

// ---------------- kernel 0: f32 -> bf16 convert ----------------
__global__ __launch_bounds__(256) void cvt_bf16_kernel(const float* __restrict__ s,
                                                       __bf16* __restrict__ d, int n) {
    int i = blockIdx.x * 256 + threadIdx.x;
    if (i < n) d[i] = to_bf16(s[i]);
}

// ---------------- kernel 1: conv 1x1 GEMM via WMMA ----------------
// img[b][m][p] = sum_c conv_w[m][c]*image[b][c][p] + conv_b[m]
// writes c-major img and p-major imgT. Double-buffered LDS B-tile.
__device__ __forceinline__ void store_conv_tile(v8f c, const float* __restrict__ conv_b,
                                                float* __restrict__ img, float* __restrict__ imgT,
                                                int b, int m0, int p, int hi) {
#pragma unroll
    for (int j = 0; j < 8; ++j) c[j] += conv_b[m0 + hi + j];
    if (p < PSP) {
        float* tp = imgT + ((size_t)b * PSP + p) * NFC + m0 + hi;
        v4f lo4 = { c[0], c[1], c[2], c[3] };
        v4f hi4 = { c[4], c[5], c[6], c[7] };
        *(v4f*)tp       = lo4;
        *(v4f*)(tp + 4) = hi4;
#pragma unroll
        for (int j = 0; j < 8; ++j)
            img[((size_t)(b * NFC + m0 + hi + j)) * PSP + p] = c[j];
    }
}

__global__ __launch_bounds__(256) void conv_gemm_kernel(const float* __restrict__ image,
                                                        const __bf16* __restrict__ cwb,
                                                        const float* __restrict__ conv_b,
                                                        float* __restrict__ img,
                                                        float* __restrict__ imgT) {
    __shared__ __align__(16) __bf16 Bs[2][16 * 40];   // ping-pong [n][k], k padded 32->40
    int b  = blockIdx.y;
    int p0 = blockIdx.x * 16;
    int tid = threadIdx.x, wave = tid >> 5, lane = tid & 31;
    int m0a = wave * 32, m0b = wave * 32 + 16;
    v8f c0 = {}, c1 = {};

    // staging role of this thread: column n, k-rows k and k+16 of the 32x16 tile
    int ns = tid & 15;
    int ks = tid >> 4;                 // 0..15
    int p  = p0 + ns;
    int pc = (p < PSP) ? p : (PSP - 1);     // clamped (branchless OOB)
    bool pv = (p < PSP);
    const float* ib = image + (size_t)b * DCH * PSP + pc;   // + k*PSP indexes channel k

    // prologue: stage tile 0
    {
        float v0 = __builtin_nontemporal_load(ib + (size_t)ks * PSP);
        float v1 = __builtin_nontemporal_load(ib + (size_t)(ks + 16) * PSP);
        if (!pv) { v0 = 0.f; v1 = 0.f; }
        Bs[0][ns * 40 + ks]      = to_bf16(v0);
        Bs[0][ns * 40 + ks + 16] = to_bf16(v1);
    }
    __syncthreads();

    for (int kt = 0; kt < KSTEPS; ++kt) {
        int cur = kt & 1, nxt = cur ^ 1;
        int k0 = kt * 32;

        // issue next tile's global loads first (overlap with WMMA)
        float n0 = 0.f, n1 = 0.f;
        if (kt + 1 < KSTEPS) {
            int kk = k0 + 32 + ks;
            n0 = __builtin_nontemporal_load(ib + (size_t)kk * PSP);
            n1 = __builtin_nontemporal_load(ib + (size_t)(kk + 16) * PSP);
            if (!pv) { n0 = 0.f; n1 = 0.f; }
        }

        // compute from current LDS buffer
        const __bf16* bp = &Bs[cur][(lane & 15) * 40 + ((lane & 16) ? 16 : 0)];
        v16bf bf = cat16(*(const v8bf*)bp, *(const v8bf*)(bp + 8));
        v16bf a0 = load_a_frag(cwb, m0a + (lane & 15), DCH, k0, lane);
        v16bf a1 = load_a_frag(cwb, m0b + (lane & 15), DCH, k0, lane);
        c0 = wmma_bf16(a0, bf, c0);
        c1 = wmma_bf16(a1, bf, c1);

        // convert + store next tile into the other buffer
        if (kt + 1 < KSTEPS) {
            Bs[nxt][ns * 40 + ks]      = to_bf16(n0);
            Bs[nxt][ns * 40 + ks + 16] = to_bf16(n1);
        }
        __syncthreads();
    }

    int pp = p0 + (lane & 15);
    int hi = (lane & 16) ? 8 : 0;
    store_conv_tile(c0, conv_b, img, imgT, b, m0a, pp, hi);
    store_conv_tile(c1, conv_b, img, imgT, b, m0b, pp, hi);
}

// ---------------- generic X @ W^T + bias GEMM via WMMA ----------------
template<bool RELU>
__global__ __launch_bounds__(256) void gemm_bt_kernel(const __bf16* __restrict__ A, int rsA,
                                                      const __bf16* __restrict__ W, int rsW,
                                                      const float* __restrict__ bias, int K,
                                                      int Mtiles, int Ntiles,
                                                      float* __restrict__ outF,
                                                      __bf16* __restrict__ outB, int ldo) {
    int wave = threadIdx.x >> 5, lane = threadIdx.x & 31;
    int tile = blockIdx.x * 8 + wave;
    if (tile >= Mtiles * Ntiles) return;
    int m0 = (tile / Ntiles) * 16;
    int n0 = (tile % Ntiles) * 16;
    v8f c = {};
    for (int k0 = 0; k0 < K; k0 += 32) {
        v16bf a = load_a_frag(A, m0 + (lane & 15), rsA, k0, lane);
        v16bf b = load_bt_frag(W, n0, rsW, k0, lane);
        c = wmma_bf16(a, b, c);
    }
    int col   = n0 + (lane & 15);
    float bv  = bias[col];
    int mBase = m0 + ((lane & 16) ? 8 : 0);
#pragma unroll
    for (int j = 0; j < 8; ++j) {
        float v = c[j] + bv;
        if (RELU) v = fmaxf(v, 0.f);
        size_t o = (size_t)(mBase + j) * ldo + col;
        if (outF) outF[o] = v;
        if (outB) outB[o] = to_bf16(v);
    }
}

// ---------------- attention: logits ----------------
// logits[b][n][p] = sum_c emb[b][n][c] * img[b][c][p]
__global__ __launch_bounds__(256) void logits_kernel(const float* __restrict__ img,
                                                     const float* __restrict__ embf,
                                                     float* __restrict__ logits) {
    int b  = blockIdx.y;
    int p0 = blockIdx.x * 64;
    __shared__ float es[NN * NFC];
    int tid = threadIdx.x;
    for (int i = tid; i < NN * NFC; i += 256) es[i] = embf[(size_t)b * NN * NFC + i];
    __syncthreads();
    int p = p0 + (tid & 63);
    int g = tid >> 6;            // n in {g, g+4, g+8}
    if (p >= PSP) return;
    float a0 = 0.f, a1 = 0.f, a2 = 0.f;
    for (int c = 0; c < NFC; ++c) {
        float v = img[((size_t)(b * NFC + c)) * PSP + p];
        a0 += es[g * NFC + c] * v;
        a1 += es[(g + 4) * NFC + c] * v;
        a2 += es[(g + 8) * NFC + c] * v;
    }
    logits[((size_t)(b * NN + g)) * PSP + p]     = a0;
    logits[((size_t)(b * NN + g + 4)) * PSP + p] = a1;
    logits[((size_t)(b * NN + g + 8)) * PSP + p] = a2;
}

// ---------------- attention: softmax over p (row per block) ----------------
__global__ __launch_bounds__(256) void softmax_kernel(float* __restrict__ L) {
    float* Lp = L + (size_t)blockIdx.x * PSP;
    __shared__ float red[256];
    int tid = threadIdx.x;
    float m = -3.4e38f;
    for (int p = tid; p < PSP; p += 256) m = fmaxf(m, Lp[p]);
    red[tid] = m; __syncthreads();
    for (int off = 128; off > 0; off >>= 1) {
        if (tid < off) red[tid] = fmaxf(red[tid], red[tid + off]);
        __syncthreads();
    }
    float M = red[0]; __syncthreads();
    float s = 0.f;
    for (int p = tid; p < PSP; p += 256) s += expf(Lp[p] - M);
    red[tid] = s; __syncthreads();
    for (int off = 128; off > 0; off >>= 1) {
        if (tid < off) red[tid] += red[tid + off];
        __syncthreads();
    }
    float inv = 1.f / red[0];
    for (int p = tid; p < PSP; p += 256) Lp[p] = expf(Lp[p] - M) * inv;
}

// ---------------- attention: ctx ----------------
// ctx[b][n][c] = sum_p A[b][n][p] * imgT[b][p][c]
__global__ __launch_bounds__(256) void ctx_kernel(const float* __restrict__ imgT,
                                                  const float* __restrict__ A,
                                                  float* __restrict__ ctx) {
    int b  = blockIdx.y;
    int c0 = blockIdx.x * 64;
    __shared__ float as[NN * 128];
    int tid = threadIdx.x;
    int c = c0 + (tid & 63);
    int g = tid >> 6;
    float a0 = 0.f, a1 = 0.f, a2 = 0.f;
    for (int pb = 0; pb < PSP; pb += 128) {
        int pmax = min(128, PSP - pb);
        __syncthreads();
        for (int i = tid; i < NN * 128; i += 256) {
            int n = i >> 7, pp = i & 127;
            as[i] = (pp < pmax) ? A[((size_t)(b * NN + n)) * PSP + pb + pp] : 0.f;
        }
        __syncthreads();
        for (int pp = 0; pp < pmax; ++pp) {
            float v = imgT[((size_t)b * PSP + pb + pp) * NFC + c];
            a0 += as[g * 128 + pp] * v;
            a1 += as[(g + 4) * 128 + pp] * v;
            a2 += as[(g + 8) * 128 + pp] * v;
        }
    }
    ctx[((size_t)(b * NN + g)) * NFC + c]     = a0;
    ctx[((size_t)(b * NN + g + 4)) * NFC + c] = a1;
    ctx[((size_t)(b * NN + g + 8)) * NFC + c] = a2;
}

// ---------------- layernorm( A + B ) ----------------
__global__ __launch_bounds__(256) void ln_add_kernel(const float* __restrict__ A,
                                                     const float* __restrict__ B,
                                                     const float* __restrict__ g,
                                                     const float* __restrict__ bb,
                                                     float* __restrict__ outF,
                                                     __bf16* __restrict__ outB) {
    __shared__ float s1[256], s2[256];
    int r = blockIdx.x, tid = threadIdx.x;
    size_t idx = (size_t)r * NFC + tid;
    float v = A[idx] + B[idx];
    s1[tid] = v; s2[tid] = v * v; __syncthreads();
    for (int off = 128; off > 0; off >>= 1) {
        if (tid < off) { s1[tid] += s1[tid + off]; s2[tid] += s2[tid + off]; }
        __syncthreads();
    }
    float mean = s1[0] * (1.f / NFC);
    float var  = s2[0] * (1.f / NFC) - mean * mean;
    float rs   = rsqrtf(var + 1e-5f);
    float o = (v - mean) * rs * g[tid] + bb[tid];
    if (outF) outF[idx] = o;
    if (outB) outB[idx] = to_bf16(o);
}

// ---------------- launch ----------------
extern "C" void kernel_launch(void* const* d_in, const int* in_sizes, int n_in,
                              void* d_out, int out_size, void* d_ws, size_t ws_size,
                              hipStream_t stream) {
    (void)in_sizes; (void)n_in; (void)out_size; (void)ws_size;
    const float* roi    = (const float*)d_in[0];   // [384,1024]
    const float* image  = (const float*)d_in[1];   // [32,768,28,43]
    const float* conv_w = (const float*)d_in[2];   // [256,768]
    const float* conv_b = (const float*)d_in[3];
    const float* emb_w  = (const float*)d_in[4];   // [256,1024]
    const float* emb_b  = (const float*)d_in[5];
    const float* ln1_g  = (const float*)d_in[6];
    const float* ln1_b  = (const float*)d_in[7];
    const float* ffn_w1 = (const float*)d_in[8];   // [256,256]
    const float* ffn_b1 = (const float*)d_in[9];
    const float* ffn_w2 = (const float*)d_in[10];  // [256,256]
    const float* ffn_b2 = (const float*)d_in[11];
    const float* ln2_g  = (const float*)d_in[12];
    const float* ln2_b  = (const float*)d_in[13];
    float* outp = (float*)d_out;                   // [384,256]

    // workspace carve-up (256B aligned)
    char* w = (char*)d_ws;
    auto alloc = [&](size_t bytes) { void* p = (void*)w; w += (bytes + 255) & ~(size_t)255; return p; };
    __bf16* cwb   = (__bf16*)alloc((size_t)NFC * DCH * 2);
    __bf16* embwb = (__bf16*)alloc((size_t)NFC * NFB * 2);
    __bf16* w1b   = (__bf16*)alloc((size_t)NFC * NFC * 2);
    __bf16* w2b   = (__bf16*)alloc((size_t)NFC * NFC * 2);
    __bf16* roib  = (__bf16*)alloc((size_t)RR * NFB * 2);
    __bf16* xb    = (__bf16*)alloc((size_t)RR * NFC * 2);
    __bf16* hb    = (__bf16*)alloc((size_t)RR * NFC * 2);
    float* img    = (float*)alloc((size_t)BT * NFC * PSP * 4);
    float* imgT   = (float*)alloc((size_t)BT * PSP * NFC * 4);
    float* embf   = (float*)alloc((size_t)RR * NFC * 4);
    float* logits = (float*)alloc((size_t)BT * NN * PSP * 4);
    float* ctx    = (float*)alloc((size_t)RR * NFC * 4);
    float* xf     = (float*)alloc((size_t)RR * NFC * 4);
    float* h2     = (float*)alloc((size_t)RR * NFC * 4);

    // K0: weight / roi conversions to bf16
    auto cvt = [&](const float* s, __bf16* d, int n) {
        cvt_bf16_kernel<<<(n + 255) / 256, 256, 0, stream>>>(s, d, n);
    };
    cvt(conv_w, cwb,   NFC * DCH);
    cvt(emb_w,  embwb, NFC * NFB);
    cvt(ffn_w1, w1b,   NFC * NFC);
    cvt(ffn_w2, w2b,   NFC * NFC);
    cvt(roi,    roib,  RR * NFB);

    // K1: conv GEMM (76 p-tiles x 32 batches)
    conv_gemm_kernel<<<dim3(76, BT), 256, 0, stream>>>(image, cwb, conv_b, img, imgT);

    // K2: emb = roi @ emb_w^T + emb_b   (24x16 tiles, 8 waves/block)
    gemm_bt_kernel<false><<<48, 256, 0, stream>>>(roib, NFB, embwb, NFB, emb_b, NFB,
                                                  24, 16, embf, (__bf16*)nullptr, NFC);

    // K3: attention
    logits_kernel<<<dim3(19, BT), 256, 0, stream>>>(img, embf, logits);
    softmax_kernel<<<BT * NN, 256, 0, stream>>>(logits);
    ctx_kernel<<<dim3(4, BT), 256, 0, stream>>>(imgT, logits, ctx);

    // K4: x = LN1(ctx + emb)  -> f32 + bf16
    ln_add_kernel<<<RR, 256, 0, stream>>>(ctx, embf, ln1_g, ln1_b, xf, xb);

    // K5: h = relu(x @ w1^T + b1) -> bf16
    gemm_bt_kernel<true><<<48, 256, 0, stream>>>(xb, NFC, w1b, NFC, ffn_b1, NFC,
                                                 24, 16, (float*)nullptr, hb, NFC);

    // K6: h2 = h @ w2^T + b2 -> f32
    gemm_bt_kernel<false><<<48, 256, 0, stream>>>(hb, NFC, w2b, NFC, ffn_b2, NFC,
                                                  24, 16, h2, (__bf16*)nullptr, NFC);

    // K7: out = LN2(x + h2)
    ln_add_kernel<<<RR, 256, 0, stream>>>(xf, h2, ln2_g, ln2_b, outp, (__bf16*)nullptr);
}
